// Scenario_Robust_OPF_46926812676566
// MI455X (gfx1250) — matrix operand order repaired and checked by reading, not compiled, not run on hardware
//
#include <hip/hip_runtime.h>

// ---------------- problem dimensions ----------------
#define U_   160
#define LN_  512
#define NN_  400
#define S_   16
#define W_   64
#define NTH  1024
#define NWAVE 32
#define CVIOL 2.0e4f
#define NPDHG 500
#define NPOW  30

// ---------------- x-vector layout (floats) ----------------
#define XP   0
#define XRU  160
#define XRD  320
#define XPU  480
#define XPD  3040
#define XSU  5600
#define XSD  6000
#define XTOT 6400

// ---------------- y-vector layout (floats, within Y) ----------------
#define Y1o  0
#define Y2o  160
#define Y3o  320
#define Y4o  832
#define Y5o  1344
#define Y6o  3904
#define Y7o  6464
#define Y8o  14656
#define YZ1  22848
#define YZ2  22852
#define YTOT 22868

// ---------------- LDS layout (float offsets, all 16B-aligned) ----------------
#define L_X     0
#define L_XN    6400
#define L_XB    12800
#define L_Y     19200
#define L_FP    42068
#define L_PN    42580
#define L_T3    43092
#define L_S78S  43604
#define L_SCEN  44116
#define L_HGT   52308
#define L_PT    54868
#define L_G160  55268
#define L_XTB   55428   /* 2560 bf16 = 1280 floats */
#define L_S78T  56708   /* 8192 bf16 = 4096 floats */
#define L_RED   60804
#define L_MISC  60836   /* [0]=beq1, [8..23]=e2 scratch, [32..47]=beq2 */
#define L_TOTAL 60900

// ---------------- workspace layout (float offsets) ----------------
#define WS_HG    0        /* 81920 f32 */
#define WS_HGB   81920    /* 81920 bf16 */
#define WS_HGTB  122880   /* 81920 bf16 (transposed) */
#define WS_HW    163840   /* 32768 f32 */
#define WS_CFLOW 196608   /* 512 */
#define WS_LD    197120   /* 400 (pad 512) */
#define WS_B7    197632   /* 8192 */
#define WS_B8    205824   /* 8192 */

typedef __attribute__((ext_vector_type(16))) __bf16 v16bf;
typedef __attribute__((ext_vector_type(8)))  float  v8f;

__device__ __forceinline__ unsigned short f2bf_bits(float x) {
  union { float f; unsigned u; } v; v.f = x;
  unsigned r = v.u + 0x7fffu + ((v.u >> 16) & 1u);   // round-to-nearest-even
  return (unsigned short)(r >> 16);
}

// A-fragment: 16x32 bf16, row-major source, ISA layout (8-interleaved K halves)
__device__ __forceinline__ v16bf load_a16(const unsigned short* M, int ld, int row0, int k0, int lane) {
  const int m = row0 + (lane & 15), h = (lane >> 4) & 1;
  const unsigned short* p = M + (size_t)m * ld + k0 + 8 * h;
  union { v16bf v; uint4 q[2]; } u;
  u.q[0] = *(const uint4*)p;          // K = k0+8h .. +7
  u.q[1] = *(const uint4*)(p + 16);   // K = k0+16+8h .. +7
  return u.v;
}

// B-fragment: 32x16 bf16 from BT[n][k] (transposed storage), ISA layout (8-interleaved K halves)
__device__ __forceinline__ v16bf load_b16(const unsigned short* BT, int ld, int k0, int lane) {
  const int n = lane & 15, h = (lane >> 4) & 1;
  const unsigned short* p = BT + (size_t)n * ld + k0 + 8 * h;
  union { v16bf v; uint4 q[2]; } u;
  u.q[0] = *(const uint4*)p;          // K = k0+8h .. +7
  u.q[1] = *(const uint4*)(p + 16);   // K = k0+16+8h .. +7
  return u.v;
}

__device__ __forceinline__ v8f wmma_bf16(v16bf a, v16bf b, v8f c) {
  return __builtin_amdgcn_wmma_f32_16x16x32_bf16(false, a, false, b, (short)0, c, false, false);
}

__device__ __forceinline__ float wave_red(float v) {
  #pragma unroll
  for (int off = 16; off; off >>= 1) v += __shfl_down(v, off, 32);
  return v;
}

__device__ float block_sum(float v, float* red, int tid) {
  v = wave_red(v);
  __syncthreads();
  if ((tid & 31) == 0) red[tid >> 5] = v;
  __syncthreads();
  if (tid < 32) {
    float t = red[tid];
    t = wave_red(t);
    if (tid == 0) red[0] = t;
  }
  __syncthreads();
  return red[0];
}

struct Ctx {
  float *X, *XN, *XB, *Y, *FP, *PN, *T3, *S78S, *SCEN, *HGT, *PT, *G160, *RED, *MISC;
  unsigned short *XTB, *S78T;
  const float *PTDF, *Pmax, *Cost, *Crup, *Crdn, *Cap, *Hg, *cflow, *b7, *b8;
  const unsigned short *Hgb, *HgTb;
  int tid, lane, wid;
};

// ---------------- A(x): fills/updates duals ----------------
__device__ void A_apply(const Ctx& c, const float* xv, int pdhg, float sig) {
  const int tid = c.tid, lane = c.lane, wid = c.wid;
  // Phase 0: d = su - sd ; XT = (pu-pd)^T in bf16
  for (int n = tid; n < NN_; n += NTH) c.PT[n] = xv[XSU + n] - xv[XSD + n];
  for (int i = tid; i < U_ * S_; i += NTH) {
    int u = i >> 4, s = i & 15;
    c.XTB[s * U_ + u] = f2bf_bits(xv[XPU + i] - xv[XPD + i]);
  }
  __syncthreads();
  // equality residual pieces
  float e1   = block_sum(tid < U_  ? xv[XP + tid] : 0.f, c.RED, tid);
  float susd = block_sum(tid < NN_ ? (xv[XSU + tid] - xv[XSD + tid]) : 0.f, c.RED, tid);
  if (tid < S_) {
    float t = 0.f;
    for (int u = 0; u < U_; ++u) t += xv[XPU + u * S_ + tid] - xv[XPD + u * S_ + tid];
    c.MISC[8 + tid] = t + susd;            // e2[s]
  }
  // Phase 1: fp = Hg@p, pn = PTDF@(su-sd) -- wave-per-row, lane-strided (coalesced)
  for (int t = wid; t < 2 * LN_; t += NWAVE) {
    float s = 0.f;
    if (t < LN_) {
      const float* row = c.Hg + (size_t)t * U_;
      for (int k = lane; k < U_; k += 32) s = fmaf(row[k], xv[XP + k], s);
    } else {
      const float* row = c.PTDF + (size_t)(t - LN_) * NN_;
      for (int k = lane; k < NN_; k += 32) s = fmaf(row[k], c.PT[k], s);
    }
    s = wave_red(s);
    if (lane == 0) { if (t < LN_) c.FP[t] = s; else c.PN[t - LN_] = s; }
  }
  __syncthreads();
  // Phase 2: scen = Hg@(pu-pd) + fp + pn  via WMMA bf16 (tile per wave, EXEC all-ones)
  {
    const int row0 = wid << 4;
    v8f acc = {0.f,0.f,0.f,0.f,0.f,0.f,0.f,0.f};
    #pragma unroll
    for (int k0 = 0; k0 < U_; k0 += 32) {
      v16bf a = load_a16(c.Hgb, U_, row0, k0, lane);
      v16bf b = load_b16(c.XTB, U_, k0, lane);
      acc = wmma_bf16(a, b, acc);
    }
    const int h = (lane >> 4) & 1, col = lane & 15;
    #pragma unroll
    for (int v = 0; v < 8; ++v) {
      int r = row0 + v + 8 * h;
      c.SCEN[r * S_ + col] = acc[v] + c.FP[r] + c.PN[r];
    }
  }
  __syncthreads();
  // Phase 3: dual prox update (pdhg) or raw A(x) (power iteration)
  if (pdhg) {
    for (int u = tid; u < U_; u += NTH) {
      float p = xv[XP + u];
      c.Y[Y1o+u] = fmaxf(c.Y[Y1o+u] + sig * (p + xv[XRU+u] - c.Pmax[u]), 0.f);
      c.Y[Y2o+u] = fmaxf(c.Y[Y2o+u] + sig * (xv[XRD+u] - p), 0.f);
    }
    for (int r = tid; r < LN_; r += NTH) {
      float cf = c.cflow[r], cap = c.Cap[r], f = c.FP[r];
      c.Y[Y3o+r] = fmaxf(c.Y[Y3o+r] + sig * ( f - (cap - cf)), 0.f);
      c.Y[Y4o+r] = fmaxf(c.Y[Y4o+r] + sig * (-f - (cap + cf)), 0.f);
    }
    for (int i = tid; i < U_ * S_; i += NTH) {
      int u = i >> 4;
      c.Y[Y5o+i] = fmaxf(c.Y[Y5o+i] + sig * (xv[XPU+i] - xv[XRU+u]), 0.f);
      c.Y[Y6o+i] = fmaxf(c.Y[Y6o+i] + sig * (xv[XPD+i] - xv[XRD+u]), 0.f);
    }
    for (int i = tid; i < LN_ * S_; i += NTH) {
      float sc = c.SCEN[i];
      c.Y[Y7o+i] = fmaxf(c.Y[Y7o+i] + sig * ( sc - c.b7[i]), 0.f);
      c.Y[Y8o+i] = fmaxf(c.Y[Y8o+i] + sig * (-sc - c.b8[i]), 0.f);
    }
    if (tid == 0)  c.Y[YZ1]     += sig * (e1 - c.MISC[0]);
    if (tid < S_)  c.Y[YZ2+tid] += sig * (c.MISC[8+tid] - c.MISC[32+tid]);
  } else {
    for (int u = tid; u < U_; u += NTH) {
      float p = xv[XP + u];
      c.Y[Y1o+u] = p + xv[XRU+u];
      c.Y[Y2o+u] = xv[XRD+u] - p;
    }
    for (int r = tid; r < LN_; r += NTH) {
      c.Y[Y3o+r] =  c.FP[r];
      c.Y[Y4o+r] = -c.FP[r];
    }
    for (int i = tid; i < U_ * S_; i += NTH) {
      int u = i >> 4;
      c.Y[Y5o+i] = xv[XPU+i] - xv[XRU+u];
      c.Y[Y6o+i] = xv[XPD+i] - xv[XRD+u];
    }
    for (int i = tid; i < LN_ * S_; i += NTH) {
      float sc = c.SCEN[i];
      c.Y[Y7o+i] =  sc;
      c.Y[Y8o+i] = -sc;
    }
    if (tid == 0) c.Y[YZ1]     = e1;
    if (tid < S_) c.Y[YZ2+tid] = c.MISC[8+tid];
  }
  __syncthreads();
}

// ---------------- A^T(y): primal gradient + prox ----------------
__device__ void AT_apply(const Ctx& c, const float* xv, float* xo, int pdhg, float tau) {
  const int tid = c.tid, lane = c.lane, wid = c.wid;
  // Phase 0: s78 = y7-y8 (bf16 transposed), s78s row sums, t3 = y3-y4+s78s
  for (int r = tid; r < LN_; r += NTH) {
    float acc = 0.f;
    #pragma unroll
    for (int s = 0; s < S_; ++s) {
      float d = c.Y[Y7o + r * S_ + s] - c.Y[Y8o + r * S_ + s];
      c.S78T[s * LN_ + r] = f2bf_bits(d);
      acc += d;
    }
    c.S78S[r] = acc;
    c.T3[r]   = c.Y[Y3o + r] - c.Y[Y4o + r] + acc;
  }
  __syncthreads();
  // Phase 1 (concurrent roles):
  //   waves 0..9  : hgt = Hg^T @ s78 via WMMA (10 tiles x 16 K-steps)
  //   tid 320..719: pt[n]   = PTDF^T @ s78s  (coalesced column dot)
  //   tid 720..879: g160[u] = Hg^T @ t3      (coalesced column dot)
  if (tid < 320) {
    const int row0 = wid << 4;
    v8f acc = {0.f,0.f,0.f,0.f,0.f,0.f,0.f,0.f};
    for (int k0 = 0; k0 < LN_; k0 += 32) {
      v16bf a = load_a16(c.HgTb, LN_, row0, k0, lane);
      v16bf b = load_b16(c.S78T, LN_, k0, lane);
      acc = wmma_bf16(a, b, acc);
    }
    const int h = (lane >> 4) & 1, col = lane & 15;
    #pragma unroll
    for (int v = 0; v < 8; ++v)
      c.HGT[(row0 + v + 8 * h) * S_ + col] = acc[v];
  } else if (tid < 720) {
    const int n = tid - 320;
    float s = 0.f;
    for (int r = 0; r < LN_; ++r) s = fmaf(c.PTDF[(size_t)r * NN_ + n], c.S78S[r], s);
    c.PT[n] = s;
  } else if (tid < 880) {
    const int u = tid - 720;
    float s = 0.f;
    for (int r = 0; r < LN_; ++r) s = fmaf(c.Hg[(size_t)r * U_ + u], c.T3[r], s);
    c.G160[u] = s;
  }
  __syncthreads();
  // Phase 2: primal update
  float z1 = c.Y[YZ1];
  float zs = 0.f;
  #pragma unroll
  for (int s = 0; s < S_; ++s) zs += c.Y[YZ2 + s];
  for (int u = tid; u < U_; u += NTH) {
    float y1 = c.Y[Y1o+u], y2 = c.Y[Y2o+u];
    float s5 = 0.f, s6 = 0.f;
    #pragma unroll
    for (int s = 0; s < S_; ++s) { s5 += c.Y[Y5o + u*S_ + s]; s6 += c.Y[Y6o + u*S_ + s]; }
    float gp  = y1 - y2 + c.G160[u] + z1;
    float gru = y1 - s5;
    float grd = y2 - s6;
    if (pdhg) {
      xo[XP +u] = fmaxf(xv[XP +u] - tau * (c.Cost[u] + gp ), 0.f);
      xo[XRU+u] = fmaxf(xv[XRU+u] - tau * (c.Crup[u] + gru), 0.f);
      xo[XRD+u] = fmaxf(xv[XRD+u] - tau * (c.Crdn[u] + grd), 0.f);
    } else { xo[XP+u] = gp; xo[XRU+u] = gru; xo[XRD+u] = grd; }
  }
  for (int i = tid; i < U_ * S_; i += NTH) {
    int s = i & 15;
    float z2 = c.Y[YZ2 + s], h = c.HGT[i];
    float gpu = c.Y[Y5o+i] + h + z2;
    float gpd = c.Y[Y6o+i] - h - z2;
    if (pdhg) {
      xo[XPU+i] = fmaxf(xv[XPU+i] - tau * gpu, 0.f);
      xo[XPD+i] = fmaxf(xv[XPD+i] - tau * gpd, 0.f);
    } else { xo[XPU+i] = gpu; xo[XPD+i] = gpd; }
  }
  for (int n = tid; n < NN_; n += NTH) {
    float g = c.PT[n] + zs;
    if (pdhg) {
      xo[XSU+n] = fmaxf(xv[XSU+n] - tau * (CVIOL + g), 0.f);
      xo[XSD+n] = fmaxf(xv[XSD+n] - tau * (CVIOL - g), 0.f);
    } else { xo[XSU+n] = g; xo[XSD+n] = -g; }
  }
  __syncthreads();
}

// ---------------- setup kernels ----------------
__global__ void k_hg(const float* __restrict__ PTDF, const float* __restrict__ nodeG,
                     float* __restrict__ Hg, unsigned short* __restrict__ Hgb,
                     unsigned short* __restrict__ HgTb) {
  int i = blockIdx.x * blockDim.x + threadIdx.x;
  if (i >= LN_ * U_) return;
  int r = i / U_, u = i - r * U_;
  const float* prow = PTDF + (size_t)r * NN_;
  float s = 0.f;
  for (int n = 0; n < NN_; ++n) s = fmaf(prow[n], nodeG[n * U_ + u], s);
  Hg[i] = s;
  unsigned short b = f2bf_bits(s);
  Hgb[i] = b;
  HgTb[(size_t)u * LN_ + r] = b;
}

__global__ void k_hw(const float* __restrict__ PTDF, const float* __restrict__ nodeW,
                     float* __restrict__ Hw) {
  int i = blockIdx.x * blockDim.x + threadIdx.x;
  if (i >= LN_ * W_) return;
  int r = i / W_, w = i - r * W_;
  const float* prow = PTDF + (size_t)r * NN_;
  float s = 0.f;
  for (int n = 0; n < NN_; ++n) s = fmaf(prow[n], nodeW[n * W_ + w], s);
  Hw[i] = s;
}

__global__ void k_ld(const float* __restrict__ nodeL, const float* __restrict__ Pd,
                     float* __restrict__ Ld) {
  int n = blockIdx.x * blockDim.x + threadIdx.x;
  if (n >= NN_) return;
  float s = 0.f;
  for (int m = 0; m < NN_; ++m) s = fmaf(nodeL[(size_t)n * NN_ + m], Pd[m], s);
  Ld[n] = s;
}

__global__ void k_cflow(const float* __restrict__ Hw, const float* __restrict__ PTDF,
                        const float* __restrict__ wexp, const float* __restrict__ Ld,
                        float* __restrict__ cflow) {
  int r = blockIdx.x * blockDim.x + threadIdx.x;
  if (r >= LN_) return;
  float s = 0.f;
  for (int w = 0; w < W_; ++w) s = fmaf(Hw[(size_t)r * W_ + w], wexp[w], s);
  for (int n = 0; n < NN_; ++n) s = fmaf(-PTDF[(size_t)r * NN_ + n], Ld[n], s);
  cflow[r] = s;
}

__global__ void k_b78(const float* __restrict__ Hw, const float* __restrict__ wscen,
                      const float* __restrict__ Cap, const float* __restrict__ cflow,
                      float* __restrict__ b7, float* __restrict__ b8) {
  int i = blockIdx.x * blockDim.x + threadIdx.x;
  if (i >= LN_ * S_) return;
  int r = i / S_, s = i - r * S_;
  float bw = 0.f;
  for (int w = 0; w < W_; ++w) bw = fmaf(Hw[(size_t)r * W_ + w], wscen[s * W_ + w], bw);
  b7[i] = Cap[r] - cflow[r] - bw;
  b8[i] = Cap[r] + cflow[r] + bw;
}

// ---------------- persistent solver kernel (single WGP) ----------------
extern "C" __global__ __launch_bounds__(NTH, 1)
void opf_main(const float* __restrict__ wscen, const float* __restrict__ Pmax,
              const float* __restrict__ Cost,  const float* __restrict__ Crup,
              const float* __restrict__ Crdn,  const float* __restrict__ PTDF,
              const float* __restrict__ Pd,    const float* __restrict__ wexp,
              const float* __restrict__ Cap,   const float* __restrict__ Hg,
              const unsigned short* __restrict__ Hgb,
              const unsigned short* __restrict__ HgTb,
              const float* __restrict__ cflow, const float* __restrict__ b7,
              const float* __restrict__ b8,    float* __restrict__ out) {
  extern __shared__ float smem[];
  Ctx c;
  c.X = smem + L_X;   c.XN = smem + L_XN; c.XB = smem + L_XB; c.Y = smem + L_Y;
  c.FP = smem + L_FP; c.PN = smem + L_PN; c.T3 = smem + L_T3; c.S78S = smem + L_S78S;
  c.SCEN = smem + L_SCEN; c.HGT = smem + L_HGT; c.PT = smem + L_PT; c.G160 = smem + L_G160;
  c.RED = smem + L_RED; c.MISC = smem + L_MISC;
  c.XTB  = (unsigned short*)(smem + L_XTB);
  c.S78T = (unsigned short*)(smem + L_S78T);
  c.PTDF = PTDF; c.Pmax = Pmax; c.Cost = Cost; c.Crup = Crup; c.Crdn = Crdn;
  c.Cap = Cap; c.Hg = Hg; c.cflow = cflow; c.b7 = b7; c.b8 = b8;
  c.Hgb = Hgb; c.HgTb = HgTb;
  c.tid = threadIdx.x; c.lane = c.tid & 31; c.wid = c.tid >> 5;
  const int tid = c.tid;

  // equality RHS: beq1 = sum(Pd) - sum(w_exp) ; beq2[s] = -sum_w w_scen[s]
  float vb = 0.f;
  if (tid < NN_) vb += Pd[tid];
  if (tid < W_)  vb -= wexp[tid];
  float beq1 = block_sum(vb, c.RED, tid);
  if (tid == 0) c.MISC[0] = beq1;
  if (tid < S_) {
    float t = 0.f;
    for (int w = 0; w < W_; ++w) t += wscen[tid * W_ + w];
    c.MISC[32 + tid] = -t;
  }

  // ---- power iteration for ||A|| ----
  for (int i = tid; i < XTOT; i += NTH) c.X[i] = 1.f;
  __syncthreads();
  for (int it = 0; it < NPOW; ++it) {
    A_apply (c, c.X, 0, 0.f);
    AT_apply(c, c.X, c.XN, 0, 0.f);
    float ss = 0.f;
    for (int i = tid; i < XTOT; i += NTH) { float v = c.XN[i]; ss = fmaf(v, v, ss); }
    float n2 = block_sum(ss, c.RED, tid);
    float inv = 1.f / sqrtf(n2);
    for (int i = tid; i < XTOT; i += NTH) c.X[i] = c.XN[i] * inv;
    __syncthreads();
  }
  A_apply (c, c.X, 0, 0.f);
  AT_apply(c, c.X, c.XN, 0, 0.f);
  float ss = 0.f;
  for (int i = tid; i < XTOT; i += NTH) { float v = c.XN[i]; ss = fmaf(v, v, ss); }
  float n2  = block_sum(ss, c.RED, tid);
  float Lop = sqrtf(sqrtf(n2));
  float tau = 0.9f / Lop, sig = tau;

  // ---- PDHG main loop ----
  for (int i = tid; i < XTOT; i += NTH) { c.X[i] = 0.f; c.XN[i] = 0.f; }
  for (int i = tid; i < YTOT; i += NTH) c.Y[i] = 0.f;
  __syncthreads();
  float* pX = c.X;
  float* pXN = c.XN;
  for (int it = 0; it < NPDHG; ++it) {
    AT_apply(c, pX, pXN, 1, tau);                 // xn = prox(x - tau(c + A^T y))
    for (int i = tid; i < XTOT; i += NTH) c.XB[i] = 2.f * pXN[i] - pX[i];
    __syncthreads();
    A_apply(c, c.XB, 1, sig);                     // y  = prox(y + sig(A xb - b))
    float* t = pX; pX = pXN; pXN = t;
  }

  // ---- outputs: p,ru,rd,pu,pd | f_up | f_down | DA_cost ----
  for (int i = tid; i < 5600; i += NTH) out[i] = pX[i];
  for (int t = c.wid; t < LN_; t += NWAVE) {
    float s = 0.f;
    const float* row = Hg + (size_t)t * U_;
    for (int k = c.lane; k < U_; k += 32) s = fmaf(row[k], pX[k], s);
    s = wave_red(s);
    if (c.lane == 0) {
      float fl = s + cflow[t];
      out[5600 + t] = Cap[t] - fl;
      out[6112 + t] = Cap[t] + fl;
    }
  }
  float dv = 0.f;
  if (tid < U_)  dv += Cost[tid]*pX[XP+tid] + Crup[tid]*pX[XRU+tid] + Crdn[tid]*pX[XRD+tid];
  if (tid < NN_) dv += CVIOL * (pX[XSU+tid] + pX[XSD+tid]);
  float cost = block_sum(dv, c.RED, tid);
  if (tid == 0) out[6624] = cost;
}

// ---------------- host launch ----------------
extern "C" void kernel_launch(void* const* d_in, const int* in_sizes, int n_in,
                              void* d_out, int out_size, void* d_ws, size_t ws_size,
                              hipStream_t stream) {
  (void)in_sizes; (void)n_in; (void)out_size; (void)ws_size;
  const float* wscen = (const float*)d_in[0];
  const float* Pmax  = (const float*)d_in[1];
  const float* Cost  = (const float*)d_in[2];
  const float* Crup  = (const float*)d_in[3];
  const float* Crdn  = (const float*)d_in[4];
  const float* PTDF  = (const float*)d_in[5];
  const float* nodeG = (const float*)d_in[6];
  const float* nodeW = (const float*)d_in[7];
  const float* nodeL = (const float*)d_in[8];
  const float* Pd    = (const float*)d_in[9];
  const float* wexp  = (const float*)d_in[10];
  const float* Cap   = (const float*)d_in[11];

  float* ws = (float*)d_ws;
  float*          Hg    = ws + WS_HG;
  unsigned short* Hgb   = (unsigned short*)(ws + WS_HGB);
  unsigned short* HgTb  = (unsigned short*)(ws + WS_HGTB);
  float*          Hw    = ws + WS_HW;
  float*          cflow = ws + WS_CFLOW;
  float*          Ld    = ws + WS_LD;
  float*          b7    = ws + WS_B7;
  float*          b8    = ws + WS_B8;

  k_hg   <<<(LN_*U_ + 255)/256, 256, 0, stream>>>(PTDF, nodeG, Hg, Hgb, HgTb);
  k_hw   <<<(LN_*W_ + 255)/256, 256, 0, stream>>>(PTDF, nodeW, Hw);
  k_ld   <<<(NN_   + 255)/256, 256, 0, stream>>>(nodeL, Pd, Ld);
  k_cflow<<<(LN_   + 255)/256, 256, 0, stream>>>(Hw, PTDF, wexp, Ld, cflow);
  k_b78  <<<(LN_*S_ + 255)/256, 256, 0, stream>>>(Hw, wscen, Cap, cflow, b7, b8);

  const size_t shmem = (size_t)L_TOTAL * sizeof(float);   // ~238 KB of WGP's 320 KB LDS
  opf_main<<<1, NTH, shmem, stream>>>(wscen, Pmax, Cost, Crup, Crdn, PTDF, Pd, wexp,
                                      Cap, Hg, Hgb, HgTb, cflow, b7, b8, (float*)d_out);
}
